// JambaModel_53919019434506
// MI455X (gfx1250) — compile-verified, matrix operand order checked by reading, not run on hardware
//
#include <hip/hip_runtime.h>
#include <hip/hip_bf16.h>
#include <math.h>
#include <stdint.h>

typedef __attribute__((ext_vector_type(16))) _Float16 v16h;
typedef __attribute__((ext_vector_type(8)))  float    v8f;

// ---------------- model dims ----------------
#define DMODEL 768
#define SEQ    1024
#define BATCH  2
#define NTOK   2048          // BATCH*SEQ
#define DINNER 1536
#define DFF    3072
#define DSTATE 16
#define DTRANK 48
#define NHEADS 12
#define HDIM   64
#define NEXP   8

// ---------------- GEMM tile ----------------
#define BM 128
#define BN 64
#define BK 32
#define APAD 36   // f32 row stride for A tiles in LDS (bank-conflict-free frag reads)
#define BPAD 68   // f32 row stride for B tiles in LDS

__device__ __forceinline__ float act_silu(float v){ return v / (1.f + expf(-v)); }

template<int ACT>
__device__ __forceinline__ float activate(float v){
    if (ACT == 1) { // silu
        return v / (1.f + expf(-v));
    } else if (ACT == 2) { // gelu (tanh approx, jax default)
        float c = 0.7978845608028654f * (v + 0.044715f * v * v * v);
        return 0.5f * v * (1.f + tanhf(c));
    } else if (ACT == 3) { // softplus
        return (v > 20.f) ? v : log1pf(expf(v));
    }
    return v;
}

// Async global->LDS copy of 16 bytes per lane (CDNA5 ASYNCcnt path).
__device__ __forceinline__ void async_cp16(const float* gp, void* lp) {
    asm volatile("global_load_async_to_lds_b128 %0, %1, off"
                 :: "v"((unsigned)(uintptr_t)lp), "v"(gp)
                 : "memory");
}
__device__ __forceinline__ void wait_async0() {
    asm volatile("s_wait_asynccnt 0x0" ::: "memory");
}
__device__ __forceinline__ void wait_async6() {
    asm volatile("s_wait_asynccnt 0x6" ::: "memory");
}

// C[M,N] = epilogue( A[M,K](lda) x B[K,N] )
// epilogue: +bias[n] -> activation -> *rowscale[m] -> (+= C if ACC)
// Double-buffered async-to-LDS staging; f32 in LDS, convert to f16 at frag build.
template<int ACT, bool ACC>
__global__ __launch_bounds__(256)
void gemm_wmma(const float* __restrict__ A, int lda,
               const float* __restrict__ B,
               float* __restrict__ C,
               const float* __restrict__ bias,
               const float* __restrict__ rowscale,
               int M, int N, int K)
{
    __shared__ float Asf[2][BM][APAD];
    __shared__ float Bsf[2][BK][BPAD];

    const int tid  = threadIdx.x;
    const int lane = tid & 31;
    const int wv   = tid >> 5;       // 0..7
    const int lh   = lane >> 4;      // lane half
    const int l16  = lane & 15;
    const int wm   = wv & 3;         // 4 wave rows
    const int wn   = wv >> 2;        // 2 wave cols
    const int m0   = blockIdx.y * BM;
    const int n0   = blockIdx.x * BN;
    const int T    = (K + BK - 1) / BK;

    // Issue one tile's async copies: uniform 6 wave-instructions per wave
    // (4 for the 128x32 A tile, 2 for the 32x64 B tile).
    auto issue_tile = [&](int t, int buf) {
        int k0 = t * BK;
        #pragma unroll
        for (int j = 0; j < 4; ++j) {                 // A tile: 4096 floats
            int f = (j * 256 + tid) * 4;
            int r = f >> 5, c = f & 31;
            int gk = k0 + c;
            if (gk >= K) gk = K - 4;                  // K%4==0 -> aligned clamp
            const float* gp = A + (size_t)(m0 + r) * lda + gk;
            async_cp16(gp, &Asf[buf][r][c]);
        }
        #pragma unroll
        for (int j = 0; j < 2; ++j) {                 // B tile: 2048 floats
            int f = (j * 256 + tid) * 4;
            int r = f >> 6, c = f & 63;
            int gk = k0 + r;
            if (gk >= K) gk = K - 1;
            int gn = n0 + c;
            if (gn >= N) gn = N - 4;                  // N%4==0 -> aligned clamp
            const float* gp = B + (size_t)gk * N + gn;
            async_cp16(gp, &Bsf[buf][r][c]);
        }
    };

    v8f acc[2][2];
    #pragma unroll
    for (int i = 0; i < 2; ++i)
        #pragma unroll
        for (int j = 0; j < 2; ++j)
            #pragma unroll
            for (int r = 0; r < 8; ++r) acc[i][j][r] = 0.f;

    issue_tile(0, 0);

    for (int t = 0; t < T; ++t) {
        int buf = t & 1;
        if (t + 1 < T) {
            issue_tile(t + 1, (t + 1) & 1);
            wait_async6();            // tile t landed; tile t+1 (6 newer) in flight
        } else {
            wait_async0();
        }
        __syncthreads();

        // K tail: zero garbage B rows (their A partners are the garbage A columns)
        int rem = K - t * BK;
        if (rem < BK) {
            for (int i = tid; i < (BK - rem) * BN; i += 256) {
                int r = rem + (i >> 6), c = i & 63;
                Bsf[buf][r][c] = 0.f;
            }
            __syncthreads();
        }

        v16h af[2], bf[2];
        #pragma unroll
        for (int ms = 0; ms < 2; ++ms) {
            int row = wm * 32 + ms * 16 + l16;
            #pragma unroll
            for (int e = 0; e < 16; ++e) {
                int kk = 8 * lh + e + ((e >= 8) ? 8 : 0);
                af[ms][e] = (_Float16)Asf[buf][row][kk];
            }
        }
        #pragma unroll
        for (int ns = 0; ns < 2; ++ns) {
            int col = wn * 32 + ns * 16 + l16;
            #pragma unroll
            for (int e = 0; e < 16; ++e) {
                int kk = 8 * lh + e + ((e >= 8) ? 8 : 0);
                bf[ns][e] = (_Float16)Bsf[buf][kk][col];
            }
        }
        #pragma unroll
        for (int ms = 0; ms < 2; ++ms)
            #pragma unroll
            for (int ns = 0; ns < 2; ++ns)
                acc[ms][ns] = __builtin_amdgcn_wmma_f32_16x16x32_f16(
                    false, af[ms], false, bf[ns], (short)0, acc[ms][ns], false, false);
        __syncthreads();   // everyone done reading buf before it is overwritten
    }

    // epilogue
    #pragma unroll
    for (int ms = 0; ms < 2; ++ms)
        #pragma unroll
        for (int ns = 0; ns < 2; ++ns)
            #pragma unroll
            for (int r = 0; r < 8; ++r) {
                int m = m0 + wm * 32 + ms * 16 + r + 8 * lh;
                int n = n0 + wn * 32 + ns * 16 + l16;
                if (n < N) {
                    float v = acc[ms][ns][r];
                    if (bias)     v += bias[n];
                    v = activate<ACT>(v);
                    if (rowscale) v *= rowscale[m];
                    size_t off = (size_t)m * N + n;
                    if (ACC) v += C[off];
                    C[off] = v;
                }
            }
}

// ---------------- RMSNorm ----------------
__global__ __launch_bounds__(256)
void rmsnorm_kernel(const float* __restrict__ in, const float* __restrict__ w,
                    float* __restrict__ out, int D)
{
    __shared__ float red[256];
    int t = blockIdx.x;
    const float* row = in + (size_t)t * D;
    float s = 0.f;
    for (int i = threadIdx.x; i < D; i += 256) { float v = row[i]; s += v * v; }
    red[threadIdx.x] = s;
    __syncthreads();
    for (int st = 128; st > 0; st >>= 1) {
        if (threadIdx.x < st) red[threadIdx.x] += red[threadIdx.x + st];
        __syncthreads();
    }
    float r = rsqrtf(red[0] / (float)D + 1e-6f);
    for (int i = threadIdx.x; i < D; i += 256)
        out[(size_t)t * D + i] = row[i] * r * w[i];
}

// ---------------- Mamba: causal depthwise conv + silu ----------------
__global__ __launch_bounds__(256)
void conv_silu_kernel(const float* __restrict__ xz, const float* __restrict__ w,
                      const float* __restrict__ bcv, float* __restrict__ out)
{
    int g = blockIdx.x * 256 + threadIdx.x;       // NTOK*DINNER
    int c = g % DINNER;
    int t = g / DINNER;
    int s = t % SEQ, b = t / SEQ;
    float acc = bcv[c];
    #pragma unroll
    for (int k = 0; k < 4; ++k) {
        int sp = s - 3 + k;
        if (sp >= 0)
            acc += xz[((size_t)(b * SEQ + sp)) * (2 * DINNER) + c] * w[c * 4 + k];
    }
    out[(size_t)t * DINNER + c] = act_silu(acc);
}

// ---------------- Mamba: selective scan ----------------
__global__ __launch_bounds__(256)
void scan_kernel(const float* __restrict__ dt, const float* __restrict__ xc,
                 const float* __restrict__ dbc, const float* __restrict__ A_log,
                 const float* __restrict__ Dvec, float* __restrict__ y)
{
    int g = blockIdx.x * 256 + threadIdx.x;       // BATCH*DINNER*DSTATE
    int n = g & 15;
    int c = (g >> 4) % DINNER;
    int b = g / (16 * DINNER);
    float A  = -expf(A_log[c * DSTATE + n]);
    float Dv = Dvec[c];
    float h = 0.f;
    size_t tb = (size_t)b * SEQ;
    for (int s = 0; s < SEQ; ++s) {
        size_t t = tb + s;
        float dtv = dt[t * DINNER + c];
        float xv  = xc[t * DINNER + c];
        float Bv  = dbc[t * 80 + DTRANK + n];
        float Cv  = dbc[t * 80 + DTRANK + DSTATE + n];
        h = expf(dtv * A) * h + dtv * Bv * xv;
        float p = h * Cv;
        p += __shfl_xor(p, 1, 16);
        p += __shfl_xor(p, 2, 16);
        p += __shfl_xor(p, 4, 16);
        p += __shfl_xor(p, 8, 16);
        if (n == 0) y[t * DINNER + c] = p + Dv * xv;
    }
}

// y *= silu(z)
__global__ __launch_bounds__(256)
void gate_kernel(const float* __restrict__ xz, float* __restrict__ y)
{
    int g = blockIdx.x * 256 + threadIdx.x;
    int c = g % DINNER;
    int t = g / DINNER;
    float z = xz[(size_t)t * (2 * DINNER) + DINNER + c];
    y[(size_t)t * DINNER + c] *= act_silu(z);
}

// ---------------- MoE router: top-2 softmax gate weights ----------------
__global__ __launch_bounds__(256)
void router_kernel(const float* __restrict__ h, const float* __restrict__ Wr,
                   float* __restrict__ we)
{
    int t = blockIdx.x * 256 + threadIdx.x;
    if (t >= NTOK) return;
    float logits[NEXP];
    #pragma unroll
    for (int e = 0; e < NEXP; ++e) logits[e] = 0.f;
    const float* row = h + (size_t)t * DMODEL;
    for (int k = 0; k < DMODEL; ++k) {
        float hv = row[k];
        #pragma unroll
        for (int e = 0; e < NEXP; ++e) logits[e] += hv * Wr[k * NEXP + e];
    }
    int i0 = 0;
    #pragma unroll
    for (int e = 1; e < NEXP; ++e) if (logits[e] > logits[i0]) i0 = e;
    int i1 = -1; float v1 = -3.0e38f;
    #pragma unroll
    for (int e = 0; e < NEXP; ++e) {
        if (e == i0) continue;
        if (logits[e] > v1) { v1 = logits[e]; i1 = e; }
    }
    float w0 = 1.f / (1.f + expf(v1 - logits[i0]));
    float w1 = 1.f - w0;
    #pragma unroll
    for (int e = 0; e < NEXP; ++e)
        we[(size_t)e * NTOK + t] = (e == i0) ? w0 : ((e == i1) ? w1 : 0.f);
}

// ---------------- Flash attention (no mask) ----------------
// q,k,v,o: [NTOK, DMODEL] f32. block = 128 threads (4 waves);
// each wave: 16 q rows; block: 64 q rows of one (b,h). K/V staged via async-to-LDS.
__global__ __launch_bounds__(128)
void attn_kernel(const float* __restrict__ q, const float* __restrict__ k,
                 const float* __restrict__ v, float* __restrict__ o)
{
    __shared__ float Ksf[32][BPAD];
    __shared__ float Vsf[32][BPAD];
    __shared__ _Float16 Ps[4][16][34];

    const int tid  = threadIdx.x;
    const int lane = tid & 31;
    const int wv   = tid >> 5;
    const int lh   = lane >> 4;
    const int l16  = lane & 15;
    const int blk  = blockIdx.x;
    const int qt   = blk & 15;              // S/64 = 16 tiles
    const int hh   = (blk >> 4) % NHEADS;
    const int b    = blk / (16 * NHEADS);
    const float scale = 0.125f;             // 1/sqrt(64)

    // Q fragments (2 head-dim chunks of 32)
    v16h aq[2];
    {
        int srow = qt * 64 + wv * 16 + l16;
        size_t qbase = ((size_t)(b * SEQ + srow)) * DMODEL + hh * HDIM;
        #pragma unroll
        for (int kc = 0; kc < 2; ++kc)
            #pragma unroll
            for (int e = 0; e < 16; ++e) {
                int d = 16 * kc + 8 * lh + e + ((e >= 8) ? 8 : 0);
                aq[kc][e] = (_Float16)(q[qbase + d] * scale);
            }
    }

    v8f oacc[4];
    float mrow[8], lrow[8];
    #pragma unroll
    for (int j = 0; j < 4; ++j)
        #pragma unroll
        for (int r = 0; r < 8; ++r) oacc[j][r] = 0.f;
    #pragma unroll
    for (int r = 0; r < 8; ++r) { mrow[r] = -3.0e38f; lrow[r] = 0.f; }

    for (int kt = 0; kt < SEQ / 32; ++kt) {
        __syncthreads();        // previous tile fully consumed
        #pragma unroll
        for (int j = 0; j < 4; ++j) {      // 2048 floats each, 16B per lane
            int f = (j * 128 + tid) * 4;
            int key = f >> 6, d = f & 63;
            size_t base = ((size_t)(b * SEQ + kt * 32 + key)) * DMODEL + hh * HDIM + d;
            async_cp16(k + base, &Ksf[key][d]);
            async_cp16(v + base, &Vsf[key][d]);
        }
        wait_async0();
        __syncthreads();

        // S tile [16 q x 32 keys]
        v8f sacc[2];
        #pragma unroll
        for (int ns = 0; ns < 2; ++ns)
            #pragma unroll
            for (int r = 0; r < 8; ++r) sacc[ns][r] = 0.f;
        #pragma unroll
        for (int ns = 0; ns < 2; ++ns) {
            int key = ns * 16 + l16;
            #pragma unroll
            for (int kc = 0; kc < 2; ++kc) {
                v16h bk;
                #pragma unroll
                for (int e = 0; e < 16; ++e) {
                    int d = 16 * kc + 8 * lh + e + ((e >= 8) ? 8 : 0);
                    bk[e] = (_Float16)Ksf[key][d];
                }
                sacc[ns] = __builtin_amdgcn_wmma_f32_16x16x32_f16(
                    false, aq[kc], false, bk, (short)0, sacc[ns], false, false);
            }
        }

        // online softmax per row (rows striped: m = r + 8*lh, cols across 16 lanes)
        #pragma unroll
        for (int r = 0; r < 8; ++r) {
            float t0 = fmaxf(sacc[0][r], sacc[1][r]);
            t0 = fmaxf(t0, __shfl_xor(t0, 1, 16));
            t0 = fmaxf(t0, __shfl_xor(t0, 2, 16));
            t0 = fmaxf(t0, __shfl_xor(t0, 4, 16));
            t0 = fmaxf(t0, __shfl_xor(t0, 8, 16));
            float newm = fmaxf(mrow[r], t0);
            float corr = expf(mrow[r] - newm);
            float p0 = expf(sacc[0][r] - newm);
            float p1 = expf(sacc[1][r] - newm);
            float ps = p0 + p1;
            ps += __shfl_xor(ps, 1, 16);
            ps += __shfl_xor(ps, 2, 16);
            ps += __shfl_xor(ps, 4, 16);
            ps += __shfl_xor(ps, 8, 16);
            lrow[r] = lrow[r] * corr + ps;
            mrow[r] = newm;
            #pragma unroll
            for (int j = 0; j < 4; ++j) oacc[j][r] *= corr;
            int prow = r + 8 * lh;
            Ps[wv][prow][l16]      = (_Float16)p0;
            Ps[wv][prow][16 + l16] = (_Float16)p1;
        }

        // P as A-fragment (16x32), V as B, accumulate O (same-wave LDS in order)
        v16h ap;
        #pragma unroll
        for (int e = 0; e < 16; ++e) {
            int kk = 8 * lh + e + ((e >= 8) ? 8 : 0);
            ap[e] = Ps[wv][l16][kk];
        }
        #pragma unroll
        for (int j = 0; j < 4; ++j) {
            v16h bv;
            int dcol = j * 16 + l16;
            #pragma unroll
            for (int e = 0; e < 16; ++e) {
                int kk = 8 * lh + e + ((e >= 8) ? 8 : 0);
                bv[e] = (_Float16)Vsf[kk][dcol];
            }
            oacc[j] = __builtin_amdgcn_wmma_f32_16x16x32_f16(
                false, ap, false, bv, (short)0, oacc[j], false, false);
        }
    }

    #pragma unroll
    for (int j = 0; j < 4; ++j)
        #pragma unroll
        for (int r = 0; r < 8; ++r) {
            int sm = qt * 64 + wv * 16 + r + 8 * lh;
            int d  = j * 16 + l16;
            o[((size_t)(b * SEQ + sm)) * DMODEL + hh * HDIM + d] = oacc[j][r] / lrow[r];
        }
}

// ---------------- host-side GEMM dispatch ----------------
static void launch_gemm(const float* A, int lda, const float* B, float* C,
                        const float* bias, const float* rowscale,
                        int M, int N, int K, int act, bool accum, hipStream_t st)
{
    dim3 grid((N + BN - 1) / BN, M / BM);
    dim3 blk(256);
    if (accum) {
        gemm_wmma<0, true><<<grid, blk, 0, st>>>(A, lda, B, C, bias, rowscale, M, N, K);
    } else {
        switch (act) {
        case 1:  gemm_wmma<1, false><<<grid, blk, 0, st>>>(A, lda, B, C, bias, rowscale, M, N, K); break;
        case 2:  gemm_wmma<2, false><<<grid, blk, 0, st>>>(A, lda, B, C, bias, rowscale, M, N, K); break;
        case 3:  gemm_wmma<3, false><<<grid, blk, 0, st>>>(A, lda, B, C, bias, rowscale, M, N, K); break;
        default: gemm_wmma<0, false><<<grid, blk, 0, st>>>(A, lda, B, C, bias, rowscale, M, N, K); break;
        }
    }
}

extern "C" void kernel_launch(void* const* d_in, const int* in_sizes, int n_in,
                              void* d_out, int out_size, void* d_ws, size_t ws_size,
                              hipStream_t stream)
{
    (void)in_sizes; (void)n_in; (void)out_size; (void)ws_size;
    float* ws = (float*)d_ws;
    size_t off = 0;
    float* xbuf  = ws + off; off += (size_t)NTOK * DMODEL;   // running residual x
    float* hbuf  = ws + off; off += (size_t)NTOK * DMODEL;   // normed activations
    float* xz    = ws + off; off += (size_t)NTOK * DFF;      // in_proj out / ffn hidden / attn O
    float* bufa  = ws + off; off += (size_t)NTOK * DINNER;   // conv out / q
    float* bufb  = ws + off; off += (size_t)NTOK * DINNER;   // dt / k
    float* bufc  = ws + off; off += (size_t)NTOK * DINNER;   // scan y / v
    float* dbc   = ws + off; off += (size_t)NTOK * 80;       // x_proj out
    float* webuf = ws + off; off += (size_t)NEXP * NTOK;     // MoE gate weights

    hipMemcpyAsync(xbuf, d_in[0], sizeof(float) * (size_t)NTOK * DMODEL,
                   hipMemcpyDeviceToDevice, stream);

    auto IN = [&](int i) { return (const float*)d_in[i]; };
    int idx = 1;

    for (int L = 0; L < 8; ++L) {
        bool is_attn = ((L + 1) % 8) == 0;
        bool use_moe = (L % 2) == 0;
        const float* norm1 = IN(idx++);
        const float* norm2 = IN(idx++);

        rmsnorm_kernel<<<NTOK, 256, 0, stream>>>(xbuf, norm1, hbuf, DMODEL);

        if (!is_attn) {
            const float* in_proj  = IN(idx++);
            const float* conv_w   = IN(idx++);
            const float* conv_b   = IN(idx++);
            const float* x_proj   = IN(idx++);
            const float* dt_proj  = IN(idx++);
            const float* dt_bias  = IN(idx++);
            const float* A_log    = IN(idx++);
            const float* Dvec     = IN(idx++);
            const float* out_proj = IN(idx++);

            launch_gemm(hbuf, DMODEL, in_proj, xz, nullptr, nullptr,
                        NTOK, 2 * DINNER, DMODEL, 0, false, stream);
            conv_silu_kernel<<<(NTOK * DINNER) / 256, 256, 0, stream>>>(xz, conv_w, conv_b, bufa);
            launch_gemm(bufa, DINNER, x_proj, dbc, nullptr, nullptr,
                        NTOK, 80, DINNER, 0, false, stream);
            launch_gemm(dbc, 80, dt_proj, bufb, dt_bias, nullptr,
                        NTOK, DINNER, DTRANK, 3 /*softplus*/, false, stream);
            scan_kernel<<<(BATCH * DINNER * DSTATE) / 256, 256, 0, stream>>>(
                bufb, bufa, dbc, A_log, Dvec, bufc);
            gate_kernel<<<(NTOK * DINNER) / 256, 256, 0, stream>>>(xz, bufc);
            launch_gemm(bufc, DINNER, out_proj, xbuf, nullptr, nullptr,
                        NTOK, DMODEL, DINNER, 0, true /*residual acc*/, stream);
        } else {
            const float* wq = IN(idx++);
            const float* wk = IN(idx++);
            const float* wvv = IN(idx++);
            const float* wo = IN(idx++);
            launch_gemm(hbuf, DMODEL, wq, bufa, nullptr, nullptr, NTOK, DMODEL, DMODEL, 0, false, stream);
            launch_gemm(hbuf, DMODEL, wk, bufb, nullptr, nullptr, NTOK, DMODEL, DMODEL, 0, false, stream);
            launch_gemm(hbuf, DMODEL, wvv, bufc, nullptr, nullptr, NTOK, DMODEL, DMODEL, 0, false, stream);
            attn_kernel<<<BATCH * NHEADS * (SEQ / 64), 128, 0, stream>>>(bufa, bufb, bufc, xz);
            launch_gemm(xz, DMODEL, wo, xbuf, nullptr, nullptr, NTOK, DMODEL, DMODEL, 0, true, stream);
        }

        rmsnorm_kernel<<<NTOK, 256, 0, stream>>>(xbuf, norm2, hbuf, DMODEL);

        if (use_moe) {
            const float* router = IN(idx++);
            const float* w1 = IN(idx++);
            const float* w2 = IN(idx++);
            router_kernel<<<NTOK / 256, 256, 0, stream>>>(hbuf, router, webuf);
            for (int e = 0; e < NEXP; ++e) {
                launch_gemm(hbuf, DMODEL, w1 + (size_t)e * DMODEL * DFF, xz,
                            nullptr, nullptr, NTOK, DFF, DMODEL, 1 /*silu*/, false, stream);
                launch_gemm(xz, DFF, w2 + (size_t)e * DFF * DMODEL, xbuf,
                            nullptr, webuf + (size_t)e * NTOK,
                            NTOK, DMODEL, DFF, 0, true /*acc*/, stream);
            }
        } else {
            const float* fw1 = IN(idx++);
            const float* fw2 = IN(idx++);
            launch_gemm(hbuf, DMODEL, fw1, xz, nullptr, nullptr,
                        NTOK, DFF, DMODEL, 2 /*gelu*/, false, stream);
            launch_gemm(xz, DFF, fw2, xbuf, nullptr, nullptr,
                        NTOK, DMODEL, DFF, 0, true, stream);
        }
    }

    const float* norm_f = IN(idx++);
    rmsnorm_kernel<<<NTOK, 256, 0, stream>>>(xbuf, norm_f, (float*)d_out, DMODEL);
}
